// PolyHashModel_87016037416980
// MI455X (gfx1250) — matrix operand order repaired and checked by reading, not compiled, not run on hardware
//
#include <hip/hip_runtime.h>

// ---------------------------------------------------------------------------
// PolyHash model for MI455X (gfx1250, wave32)
//  - WMMA bf16 16x16x32, f32 accumulate
//  - GEMM: 128x128 workgroup tile, 32x64 wave tile, double-buffered LDS fed by
//    GLOBAL_LOAD_ASYNC_TO_LDS_B128 (ASYNCcnt), one barrier per K-step
// ---------------------------------------------------------------------------

typedef __attribute__((ext_vector_type(16))) __bf16 v16bf;
typedef __attribute__((ext_vector_type(8)))  float  v8f;

union FragAB { v16bf v; uint4 q[2]; unsigned short s[16]; };

__device__ __forceinline__ unsigned short f2bf(float f) {
    unsigned int u = __float_as_uint(f);
    unsigned int r = (u + 0x7fffu + ((u >> 16) & 1u)) >> 16;   // RNE
    return (unsigned short)r;
}
__device__ __forceinline__ float bf2f(unsigned short h) {
    return __uint_as_float(((unsigned int)h) << 16);
}

// async global -> LDS copy of 16 bytes (per active lane), ASYNCcnt-tracked
__device__ __forceinline__ void async_cp16(unsigned ldsAddr, const void* g) {
    asm volatile("global_load_async_to_lds_b128 %0, %1, off"
                 :: "v"(ldsAddr), "v"((unsigned long long)(size_t)g)
                 : "memory");
}
__device__ __forceinline__ void wait_async0() {
    asm volatile("s_wait_asynccnt 0x0" ::: "memory");
}

// --------------------------- hash constants --------------------------------
__constant__ unsigned int HP[24] = {
    2654435761u,2246822519u,3266489917u,2028178513u,1220703125u,1610612741u,
    805306457u,402653189u,3674653429u,2860486313u,1073676287u,2971215073u,
    1500450271u,3267000013u,2654435789u,4049292737u,2246822531u,3266489927u,
    2028178519u,1220703133u,1610612743u,805306459u,402653191u,3674653433u};
__constant__ int PATOFF[32][3] = {
    {1,0,0},{2,0,0},{3,0,0},{4,0,0},{5,0,0},{6,0,0},{7,0,0},{8,0,0},
    {1,2,0},{2,3,0},{3,4,0},{1,3,0},{2,4,0},{1,4,0},{1,5,0},{2,5,0},
    {3,5,0},{1,6,0},{2,6,0},{1,7,0},
    {1,2,3},{1,2,4},{1,3,5},{2,3,4},{1,2,5},{1,3,4},{2,4,6},{1,4,7},
    {1,8,0},{1,9,0},{1,10,0},{1,11,0}};
__constant__ int PATLEN[32] = {1,1,1,1,1,1,1,1, 2,2,2,2,2,2,2,2,2,2,2,2,
                               3,3,3,3,3,3,3,3, 2,2,2,2};

// --------------------------- weight convert --------------------------------
__global__ __launch_bounds__(256) void cvt_bf16(const float* __restrict__ in,
                                                unsigned short* __restrict__ out,
                                                int n) {
    for (int i = blockIdx.x * blockDim.x + threadIdx.x; i < n;
         i += gridDim.x * blockDim.x)
        out[i] = f2bf(in[i]);
}

// --------------------------- feature builder -------------------------------
__global__ __launch_bounds__(256) void feature_kernel(
    const int* __restrict__ chars, const float* __restrict__ byte_embed,
    const float* __restrict__ tables, unsigned short* __restrict__ feat) {
    const int tid = threadIdx.x;
    const int tl  = tid >> 3;
    const int j   = tid & 7;
    const size_t t = (size_t)blockIdx.x * 32 + tl;
    const int s   = (int)(t & 2047);           // SEQ = 2048
    const int tok = chars[t];
    unsigned short* frow = feat + t * 512;

    const float* be = byte_embed + (size_t)tok * 256 + j * 32;
    #pragma unroll
    for (int c = 0; c < 32; c += 4) {
        float4 f = *(const float4*)(be + c);
        frow[j * 32 + c + 0] = f2bf(f.x);
        frow[j * 32 + c + 1] = f2bf(f.y);
        frow[j * 32 + c + 2] = f2bf(f.z);
        frow[j * 32 + c + 3] = f2bf(f.w);
    }
    for (int ii = 0; ii < 4; ii++) {
        const int i = j * 4 + ii;
        const int len = PATLEN[i];
        unsigned long long h = 0ull;
        for (int k = 0; k < len; k++) {
            const int o = PATOFF[i][k];
            unsigned long long tk =
                (s >= o) ? (unsigned long long)(unsigned int)chars[t - o] : 0ull;
            h ^= tk * (unsigned long long)HP[(i * 3 + k) % 24];
        }
        const unsigned long long sp = HP[(i * 3 + len) % 24];
        const long long ms = (long long)(h * sp);          // int64 wrap mul
        const unsigned long long h2 = h ^ (unsigned long long)(ms >> 16);
        const int lo = (int)(h & 4095ull);                 // floor-mod 4096
        const int hi = (int)(h2 & 4095ull);
        const float frac = (float)((h >> 3) & 255ull) * (0.4f / 255.0f);
        const float* tlo = tables + ((size_t)i * 4096 + lo) * 8;
        const float* thi = tables + ((size_t)i * 4096 + hi) * 8;
        #pragma unroll
        for (int e = 0; e < 8; e++) {
            float v = tlo[e] * (1.0f - frac) + thi[e] * frac;
            frow[256 + i * 8 + e] = f2bf(v);
        }
    }
}

// ------------------------------- GEMM --------------------------------------
// C[m][n] = sum_k X[m][k] * W[n][k]  (+bias, relu, +resid), bf16 in, f32 acc.
// 256 thr (8 waves): tile 128(M) x 128(N); wave tile 32x64 = 2x4 WMMA accs.
// Double-buffered LDS, async global->LDS staging, one barrier per K-step.
#define LDST 40   // padded LDS row stride (elements) -> 80B = 20 banks

__global__ __launch_bounds__(256) void gemm_bf16(
    const unsigned short* __restrict__ X, const unsigned short* __restrict__ W,
    const float* __restrict__ bias, const unsigned short* __restrict__ resid,
    unsigned short* __restrict__ outB, float* __restrict__ outF,
    int K, int Ncols, int doRelu) {
    __shared__ __align__(16) unsigned short At[2][128 * LDST];
    __shared__ __align__(16) unsigned short Wt[2][128 * LDST];

    const int tid  = threadIdx.x;
    const int lane = tid & 31;
    const int w    = tid >> 5;
    const int wm   = w >> 1;      // 0..3  (M 32-row block)
    const int wn   = w & 1;       // 0..1  (N 64-col block)
    const size_t m0 = (size_t)blockIdx.y * 128;
    const int    n0 = blockIdx.x * 128;

    // cooperative tile-load indices: 128 rows x 32 cols, 16 elems per thread
    const int lr = tid >> 1;            // 0..127
    const int lc = (tid & 1) * 16;      // 0 or 16

    // fragment indices (ISA 16-bit A/B layouts, wave32)
    const int rA  = lane & 15;
    const int kbA = (lane < 16) ? 0 : 8;
    const int nB  = lane & 15;
    const int kbB = (lane < 16) ? 0 : 8;

    const unsigned short* gX = X + (m0 + lr) * (size_t)K + lc;
    const unsigned short* gW = W + ((size_t)(n0 + lr)) * K + lc;
    const unsigned ldsOff = (unsigned)(lr * LDST + lc) * 2u;  // bytes
    const unsigned laBase0 = (unsigned)(unsigned long long)(void*)&At[0][0];
    const unsigned laBase1 = (unsigned)(unsigned long long)(void*)&At[1][0];
    const unsigned lwBase0 = (unsigned)(unsigned long long)(void*)&Wt[0][0];
    const unsigned lwBase1 = (unsigned)(unsigned long long)(void*)&Wt[1][0];

    v8f acc[2][4] = {};

    // prologue: stage tile 0 into buffer 0
    async_cp16(laBase0 + ldsOff,      gX);
    async_cp16(laBase0 + ldsOff + 16, gX + 8);
    async_cp16(lwBase0 + ldsOff,      gW);
    async_cp16(lwBase0 + ldsOff + 16, gW + 8);
    wait_async0();
    __syncthreads();

    const int nSteps = K >> 5;
    for (int step = 0; step < nSteps; ++step) {
        const int p = step & 1;
        // stage next tile into the other buffer (runs under the WMMAs)
        if (step + 1 < nSteps) {
            const int k0n = (step + 1) << 5;
            const unsigned la = (p ? laBase0 : laBase1) + ldsOff;
            const unsigned lw = (p ? lwBase0 : lwBase1) + ldsOff;
            async_cp16(la,      gX + k0n);
            async_cp16(la + 16, gX + k0n + 8);
            async_cp16(lw,      gW + k0n);
            async_cp16(lw + 16, gW + k0n + 8);
        }

        const unsigned short* Ab = At[p];
        const unsigned short* Wb = Wt[p];
        FragAB a[2], b[4];
        #pragma unroll
        for (int i = 0; i < 2; i++) {
            const unsigned short* ap = Ab + (wm * 32 + i * 16 + rA) * LDST + kbA;
            a[i].q[0] = *(const uint4*)(ap);        // K {kbA..kbA+7}
            a[i].q[1] = *(const uint4*)(ap + 16);   // K {kbA+16..kbA+23}
        }
        #pragma unroll
        for (int i = 0; i < 4; i++) {
            const unsigned short* bp = Wb + (wn * 64 + i * 16 + nB) * LDST + kbB;
            b[i].q[0] = *(const uint4*)(bp);        // K {kbB..kbB+7}
            b[i].q[1] = *(const uint4*)(bp + 16);   // K {kbB+16..kbB+23}
        }
        #pragma unroll
        for (int mi = 0; mi < 2; mi++)
            #pragma unroll
            for (int ni = 0; ni < 4; ni++)
                acc[mi][ni] = __builtin_amdgcn_wmma_f32_16x16x32_bf16(
                    false, a[mi].v, false, b[ni].v, (short)0, acc[mi][ni],
                    false, false);

        wait_async0();
        __syncthreads();
    }

    // epilogue: C/D layout -> lane<16: M=v, N=lane ; lane>=16: M=v+8, N=lane-16
    const int rowOff = (lane < 16) ? 0 : 8;
    const int colL   = lane & 15;
    #pragma unroll
    for (int mi = 0; mi < 2; mi++)
        #pragma unroll
        for (int ni = 0; ni < 4; ni++) {
            const int col = n0 + wn * 64 + ni * 16 + colL;
            const float bv = bias[col];
            #pragma unroll
            for (int v = 0; v < 8; v++) {
                const size_t row = m0 + wm * 32 + mi * 16 + rowOff + v;
                float val = acc[mi][ni][v] + bv;
                if (doRelu) val = fmaxf(val, 0.0f);
                if (resid)  val += bf2f(resid[row * (size_t)Ncols + col]);
                if (outB)   outB[row * (size_t)Ncols + col] = f2bf(val);
                else        outF[row * (size_t)Ncols + col] = val;
            }
        }
}

// ------------------------------ launcher -----------------------------------
extern "C" void kernel_launch(void* const* d_in, const int* in_sizes, int n_in,
                              void* d_out, int out_size, void* d_ws, size_t ws_size,
                              hipStream_t stream) {
    (void)in_sizes; (void)n_in; (void)out_size; (void)ws_size;
    const int*   chars       = (const int*)d_in[0];
    const float* byte_embed  = (const float*)d_in[1];
    const float* hash_tables = (const float*)d_in[2];
    const float* w_in        = (const float*)d_in[3];
    const float* b_in        = (const float*)d_in[4];
    const float* mlp_ws      = (const float*)d_in[5];
    const float* mlp_bs      = (const float*)d_in[6];
    const float* w_out       = (const float*)d_in[7];
    const float* b_out       = (const float*)d_in[8];

    const size_t NTOK = 65536;                       // 32 * 2048
    char* ws = (char*)d_ws;
    unsigned short* X0 = (unsigned short*)ws;                       // 64 MB
    unsigned short* X1 = (unsigned short*)(ws + NTOK * 512 * 2);    // 64 MB
    unsigned short* Wb_in  = (unsigned short*)(ws + 2 * NTOK * 512 * 2);
    unsigned short* Wb_mlp = Wb_in + 512 * 512;
    unsigned short* Wb_out = Wb_mlp + 3 * 512 * 512;

    cvt_bf16<<<512, 256, 0, stream>>>(w_in,   Wb_in,  512 * 512);
    cvt_bf16<<<512, 256, 0, stream>>>(mlp_ws, Wb_mlp, 3 * 512 * 512);
    cvt_bf16<<<512, 256, 0, stream>>>(w_out,  Wb_out, 1024 * 512);

    feature_kernel<<<NTOK / 32, 256, 0, stream>>>(chars, byte_embed,
                                                  hash_tables, X0);

    dim3 g512(4, 512), g1024(8, 512);
    // h = relu(F @ w_in^T + b_in)
    gemm_bf16<<<g512, 256, 0, stream>>>(X0, Wb_in, b_in, nullptr,
                                        X1, nullptr, 512, 512, 1);
    // 3 residual blocks: h = relu(h @ W^T + b) + h
    gemm_bf16<<<g512, 256, 0, stream>>>(X1, Wb_mlp,                 mlp_bs,
                                        X1, X0, nullptr, 512, 512, 1);
    gemm_bf16<<<g512, 256, 0, stream>>>(X0, Wb_mlp + 512 * 512,     mlp_bs + 512,
                                        X0, X1, nullptr, 512, 512, 1);
    gemm_bf16<<<g512, 256, 0, stream>>>(X1, Wb_mlp + 2 * 512 * 512, mlp_bs + 1024,
                                        X1, X0, nullptr, 512, 512, 1);
    // logits = h @ w_out^T + b_out  (f32 out)
    gemm_bf16<<<g1024, 256, 0, stream>>>(X0, Wb_out, b_out, nullptr,
                                         nullptr, (float*)d_out, 512, 1024, 0);
}
